// Int8Linear_10857677324719
// MI455X (gfx1250) — compile-verified, matrix operand order checked by reading, not run on hardware
//
#include <hip/hip_runtime.h>
#include <stdint.h>

typedef __attribute__((ext_vector_type(16))) _Float16 v16h;
typedef __attribute__((ext_vector_type(8)))  _Float16 v8h;
typedef __attribute__((ext_vector_type(2)))  _Float16 v2h;
typedef __attribute__((ext_vector_type(8)))  float    v8f;

#define M_DIM 8192
#define K_DIM 4096
#define N_DIM 11008
#define BK    64
#define LDT   72                         // padded LDS row stride in halves (144 B)
#define TILE_HALVES (128 * LDT)          // one staging tile (A or B)
#define BUF_STRIDE  (2 * TILE_HALVES)    // As+Bs per buffer

__device__ __forceinline__ v16h load_frag(const _Float16* p) {
    v8h lo = *(const v8h*)p;
    v8h hi = *(const v8h*)(p + 16);
    return __builtin_shufflevector(lo, hi,
        0, 1, 2, 3, 4, 5, 6, 7, 8, 9, 10, 11, 12, 13, 14, 15);
}

__device__ __forceinline__ int cvt2(uint32_t xs, uint32_t sel) {
    // xs = 4 sign-flipped int8; pick 2 bytes into halves {0x64,b} = 1024+u, subtract 1152 -> exact int8
    const v2h kOff = {(_Float16)1152.0f, (_Float16)1152.0f};
    uint32_t p = __builtin_amdgcn_perm(0x64646464u, xs, sel);
    v2h r = __builtin_bit_cast(v2h, p) - kOff;
    return (int)__builtin_bit_cast(uint32_t, r);
}

__device__ __forceinline__ void dequant16(int4 b, _Float16* dst) {
    uint32_t x0 = ((uint32_t)b.x) ^ 0x80808080u;
    uint32_t x1 = ((uint32_t)b.y) ^ 0x80808080u;
    uint32_t x2 = ((uint32_t)b.z) ^ 0x80808080u;
    uint32_t x3 = ((uint32_t)b.w) ^ 0x80808080u;
    int4 lo, hi;
    lo.x = cvt2(x0, 0x05010400u);
    lo.y = cvt2(x0, 0x05030402u);
    lo.z = cvt2(x1, 0x05010400u);
    lo.w = cvt2(x1, 0x05030402u);
    hi.x = cvt2(x2, 0x05010400u);
    hi.y = cvt2(x2, 0x05030402u);
    hi.z = cvt2(x3, 0x05010400u);
    hi.w = cvt2(x3, 0x05030402u);
    *(int4*)(dst + 0) = lo;
    *(int4*)(dst + 8) = hi;
}

// Async DMA: 16 B per lane, global -> LDS, no VGPR data path (ASYNCcnt-tracked).
__device__ __forceinline__ void async_copy16(const _Float16* ldsDst, const _Float16* gSrc) {
    uint32_t ldsAddr = (uint32_t)(uintptr_t)ldsDst;         // AS(3) byte offset
    uint64_t gAddr   = (uint64_t)(uintptr_t)gSrc;
    asm volatile("global_load_async_to_lds_b128 %0, %1, off"
                 :: "v"(ldsAddr), "v"(gAddr)
                 : "memory");
}

__device__ __forceinline__ void wait_async0() {
    asm volatile("s_wait_asynccnt 0x0" ::: "memory");
}

__global__ __launch_bounds__(256)
void int8_linear_wmma_f16(const _Float16* __restrict__ x,
                          const int8_t*  __restrict__ w,
                          const float*   __restrict__ scale,
                          const _Float16* __restrict__ bias,
                          float* __restrict__ out)
{
    __shared__ __attribute__((aligned(16))) _Float16 S[2 * BUF_STRIDE];

    const int m0   = blockIdx.x * 128;
    const int n0   = blockIdx.y * 128;
    const int t    = threadIdx.x;
    const int lane = t & 31;
    const int wave = t >> 5;
    const int wm   = (wave >> 2) * 64;  // wave M offset: 0 or 64
    const int wn   = (wave & 3) * 32;   // wave N offset: 0,32,64,96

    // A tile: 128 rows x 64 halves = 1024 x 16B chunks; 4 chunks/thread (async DMA).
    const int rowA = t >> 3;            // 0..31 (+32*i)
    const int colA = (t & 7) * 8;       // halves
    const _Float16* gA = x + (size_t)(m0 + rowA) * K_DIM + colA;

    // B tile: 128 rows x 64 int8 = 512 x 16B chunks; 2 chunks/thread (regs + dequant).
    const int rowB = t >> 2;            // 0..63 (+64*i)
    const int colB = (t & 3) * 16;      // bytes == k elements
    const int8_t* gB = w + (size_t)(n0 + rowB) * K_DIM + colB;

    v8f acc[4][2];
#pragma unroll
    for (int i = 0; i < 4; ++i)
#pragma unroll
        for (int j = 0; j < 2; ++j)
#pragma unroll
            for (int r = 0; r < 8; ++r)
                acc[i][j][r] = 0.0f;

    // ---- prologue: stage slab 0 into buffer 0 ----
#pragma unroll
    for (int i = 0; i < 4; ++i)
        async_copy16(&S[(rowA + 32 * i) * LDT + colA], gA + (size_t)(32 * i) * K_DIM);

    int4 bReg[2];
#pragma unroll
    for (int i = 0; i < 2; ++i) bReg[i] = *(const int4*)(gB + (size_t)(64 * i) * K_DIM);
#pragma unroll
    for (int i = 0; i < 2; ++i)
        dequant16(bReg[i], &S[TILE_HALVES + (rowB + 64 * i) * LDT + colB]);

    wait_async0();
    __syncthreads();

    // fragment lane addressing (same pattern for A and B: both (row, K) row-major)
    const int fr = lane & 15;
    const int fk = (lane >> 4) * 8;

    int bufOff = 0;
    for (int kk = 0; kk < K_DIM; kk += BK) {
        const bool hasNext = (kk + BK) < K_DIM;
        const int  nxt     = bufOff ^ BUF_STRIDE;

        // issue next slab's transfers first; compute hides the latency
        if (hasNext) {
            gA += BK; gB += BK;
#pragma unroll
            for (int i = 0; i < 4; ++i)
                async_copy16(&S[nxt + (rowA + 32 * i) * LDT + colA],
                             gA + (size_t)(32 * i) * K_DIM);
#pragma unroll
            for (int i = 0; i < 2; ++i)
                bReg[i] = *(const int4*)(gB + (size_t)(64 * i) * K_DIM);
        }

        // compute: two WMMA K-slabs from current buffer
        const _Float16* Ab = &S[bufOff];
        const _Float16* Bb = &S[bufOff + TILE_HALVES];
#pragma unroll
        for (int s = 0; s < 2; ++s) {
            const int ko = s * 32;

            v16h af[4];
#pragma unroll
            for (int i = 0; i < 4; ++i)
                af[i] = load_frag(&Ab[(wm + i * 16 + fr) * LDT + ko + fk]);

            v16h bf[2];
#pragma unroll
            for (int j = 0; j < 2; ++j)
                bf[j] = load_frag(&Bb[(wn + j * 16 + fr) * LDT + ko + fk]);

#pragma unroll
            for (int i = 0; i < 4; ++i)
#pragma unroll
                for (int j = 0; j < 2; ++j)
                    acc[i][j] = __builtin_amdgcn_wmma_f32_16x16x32_f16(
                        false, af[i], false, bf[j],
                        (short)0, acc[i][j], false, false);
        }

        // dequant + store B for next slab while WMMAs drain
        if (hasNext) {
            _Float16* Bn = &S[nxt + TILE_HALVES];
#pragma unroll
            for (int i = 0; i < 2; ++i)
                dequant16(bReg[i], &Bn[(rowB + 64 * i) * LDT + colB]);
        }

        wait_async0();     // A DMA for buf[next] complete (had the whole compute to finish)
        __syncthreads();
        bufOff = nxt;
    }

    // ---- epilogue: per-channel scale + bias, fp32 store ----
    const int rbase = (lane >> 4) * 8;
    const int cL    = lane & 15;
#pragma unroll
    for (int j = 0; j < 2; ++j) {
        const int   gn = n0 + wn + j * 16 + cL;
        const float sv = scale[gn];
        const float bv = (float)bias[gn];
#pragma unroll
        for (int i = 0; i < 4; ++i) {
            const int gm = m0 + wm + i * 16 + rbase;
#pragma unroll
            for (int r = 0; r < 8; ++r)
                out[(size_t)(gm + r) * N_DIM + gn] = fmaf(acc[i][j][r], sv, bv);
        }
    }
}

extern "C" void kernel_launch(void* const* d_in, const int* in_sizes, int n_in,
                              void* d_out, int out_size, void* d_ws, size_t ws_size,
                              hipStream_t stream) {
    const _Float16* x     = (const _Float16*)d_in[0];
    const int8_t*   w     = (const int8_t*)d_in[1];
    const float*    scale = (const float*)d_in[2];
    const _Float16* bias  = (const _Float16*)d_in[3];
    float*          out   = (float*)d_out;

    dim3 grid(M_DIM / 128, N_DIM / 128);   // (64, 86)
    int8_linear_wmma_f16<<<grid, dim3(256), 0, stream>>>(x, w, scale, bias, out);
}
